// WorkshopGNN_16793322128119
// MI455X (gfx1250) — matrix operand order repaired and checked by reading, not compiled, not run on hardware
//
#include <hip/hip_runtime.h>
#include <hip/hip_bf16.h>

#define HID 128
#define HEADS 4
#define HDIM 32
#define ENC_DIM 96
#define TYPE_DIM 32

typedef __attribute__((ext_vector_type(16))) _Float16 v16h;
typedef __attribute__((ext_vector_type(8)))  float    v8f;

__device__ __forceinline__ float wred_sum(float v) {
#pragma unroll
  for (int off = 16; off > 0; off >>= 1) v += __shfl_xor(v, off, 32);
  return v;
}
__device__ __forceinline__ unsigned fkey(float f) {
  unsigned u = __float_as_uint(f);
  return (u & 0x80000000u) ? ~u : (u | 0x80000000u);
}
__device__ __forceinline__ float fdecode(unsigned k) {
  unsigned u = (k & 0x80000000u) ? (k ^ 0x80000000u) : ~k;
  return __uint_as_float(u);
}
__device__ __forceinline__ float lrelu(float v) { return v > 0.f ? v : 0.2f * v; }

// -------- per-type encoder: Linear(din->96) -> LayerNorm -> ReLU, + type emb --------
__global__ void k_encode(const float* __restrict__ feat, int din, int nNodes, int nodeOff,
                         const float* __restrict__ W, const float* __restrict__ b,
                         const float* __restrict__ beta, const float* __restrict__ g,
                         const float* __restrict__ tembRow, float* __restrict__ x) {
  int wave = (blockIdx.x * blockDim.x + threadIdx.x) >> 5;
  int lane = threadIdx.x & 31;
  if (wave >= nNodes) return;
  const float* f = feat + (size_t)wave * din;
  float y[3];
#pragma unroll
  for (int i = 0; i < 3; ++i) {
    int c = lane + i * 32;
    float acc = b[c];
    for (int k = 0; k < din; ++k) acc += f[k] * W[k * ENC_DIM + c];
    y[i] = acc;
  }
  float mu  = wred_sum(y[0] + y[1] + y[2]) * (1.f / 96.f);
  float var = wred_sum(y[0]*y[0] + y[1]*y[1] + y[2]*y[2]) * (1.f / 96.f) - mu * mu;
  float rstd = rsqrtf(var + 1e-5f);
  float* xr = x + (size_t)(nodeOff + wave) * HID;
#pragma unroll
  for (int i = 0; i < 3; ++i) {
    int c = lane + i * 32;
    float r = (y[i] - mu) * rstd * g[c] + beta[c];
    xr[c] = r > 0.f ? r : 0.f;
  }
  xr[ENC_DIM + lane] = tembRow[lane];
}

// -------- f32 -> f16 (zero pad tail rows for the WMMA M-tiles) --------
__global__ void k_cvt(const float* __restrict__ x, _Float16* __restrict__ xh,
                      int nvalid, int ntot) {
  int i = blockIdx.x * blockDim.x + threadIdx.x;
  if (i >= ntot) return;
  xh[i] = (_Float16)(i < nvalid ? x[i] : 0.f);
}

// -------- pack W (128x128 f32 row-major, [K][N]) into WMMA B-fragment lane layout --------
// block b = kt*8+nt; within block, dword = lane*8+v; dword holds halfs
// (K = kt*32 + (lane<16?0:16) + 2v, N = nt*16 + (lane&15)) and K+1.
__global__ void k_pack_w(const float* __restrict__ W, unsigned* __restrict__ wpk) {
  int t = blockIdx.x * blockDim.x + threadIdx.x;
  if (t >= (HID * HID / 2)) return;
  int b = t >> 8, r = t & 255;
  int L = r >> 3, v = r & 7;
  int kt = b >> 3, nt = b & 7;
  int col = nt * 16 + (L & 15);
  int k0 = kt * 32 + ((L & 16) ? 16 : 0) + 2 * v;
  union { _Float16 h[2]; unsigned u; } p;
  p.h[0] = (_Float16)W[k0 * HID + col];
  p.h[1] = (_Float16)W[(k0 + 1) * HID + col];
  wpk[t] = p.u;
}

// -------- h = x_f16 @ W_f16 via v_wmma_f32_16x16x32_f16 --------
// One m-tile (16 rows) per wave; A fragments live in registers across all 8 n-tiles.
// Packed W (32KB) is staged once per block into LDS by the CDNA5 async global->LDS
// DMA (ASYNCcnt), then consumed via ds_load_b128.
__global__ void k_gemm_wmma(const _Float16* __restrict__ xh, const unsigned* __restrict__ wpk,
                            float* __restrict__ out, int mtiles) {
  __shared__ __align__(16) unsigned sB[HID * HID / 2]; // 32 KB: full packed W
  int tid = threadIdx.x;
  int wave = tid >> 5;
  int lane = tid & 31;

  // stage 32KB: 256 threads x 16B x 8 rounds, async to LDS
  {
    unsigned ldsa = (unsigned)(size_t)(&sB[0]) + (unsigned)tid * 16u; // LDS offset = addr[31:0]
    unsigned voff = (unsigned)tid * 16u;
    unsigned long long sa = (unsigned long long)wpk;
    asm volatile(
        "global_load_async_to_lds_b128 %0, %1, %2 offset:0\n\t"
        "global_load_async_to_lds_b128 %0, %1, %2 offset:4096\n\t"
        "global_load_async_to_lds_b128 %0, %1, %2 offset:8192\n\t"
        "global_load_async_to_lds_b128 %0, %1, %2 offset:12288\n\t"
        "global_load_async_to_lds_b128 %0, %1, %2 offset:16384\n\t"
        "global_load_async_to_lds_b128 %0, %1, %2 offset:20480\n\t"
        "global_load_async_to_lds_b128 %0, %1, %2 offset:24576\n\t"
        "global_load_async_to_lds_b128 %0, %1, %2 offset:28672\n\t"
        "s_wait_asynccnt 0x0"
        :: "v"(ldsa), "v"(voff), "s"(sa) : "memory");
  }
  __syncthreads();

  int mt = blockIdx.x * 8 + wave;
  if (mt < mtiles) {
    int row = mt * 16 + (lane & 15);
    int kbA = (lane & 16) ? 16 : 0; // half-offset within K window per A-layout
    union U { v16h h; uint4 u[2]; };
    U a[4];
#pragma unroll
    for (int kt = 0; kt < 4; ++kt) {
      const _Float16* pa = xh + (size_t)row * HID + kt * 32 + kbA;
      a[kt].u[0] = *(const uint4*)(pa);      // K = kb .. kb+7
      a[kt].u[1] = *(const uint4*)(pa + 8);  // K = kb+8 .. kb+15
    }
    int r0 = mt * 16 + ((lane & 16) ? 8 : 0);
#pragma unroll
    for (int nt = 0; nt < 8; ++nt) {
      v8f c;
#pragma unroll
      for (int i = 0; i < 8; ++i) c[i] = 0.f;
#pragma unroll
      for (int kt = 0; kt < 4; ++kt) {
        U b;
        const unsigned* pb = &sB[((kt << 3) + nt) * 256 + (lane << 3)];
        b.u[0] = *(const uint4*)(pb);
        b.u[1] = *(const uint4*)(pb + 4);
        c = __builtin_amdgcn_wmma_f32_16x16x32_f16(false, a[kt].h, false, b.h,
                                                   (short)0, c, false, false);
      }
      int col = (nt << 4) + (lane & 15);
#pragma unroll
      for (int i = 0; i < 8; ++i) out[(size_t)(r0 + i) * HID + col] = c[i];
    }
  }
}

// -------- per-node attention coefficients + self-loop seeded segment max --------
__global__ void k_attn_coef(const float* __restrict__ h, const float* __restrict__ asrc,
                            const float* __restrict__ adst, float* __restrict__ as_,
                            float* __restrict__ ad_, unsigned* __restrict__ segk, int N) {
  int wave = (blockIdx.x * blockDim.x + threadIdx.x) >> 5;
  int lane = threadIdx.x & 31;
  if (wave >= N) return;
  const float* hr = h + (size_t)wave * HID;
  float sv[4], dv[4];
#pragma unroll
  for (int hh = 0; hh < 4; ++hh) {
    float hv = hr[hh * 32 + lane];
    sv[hh] = wred_sum(hv * asrc[hh * 32 + lane]);
    dv[hh] = wred_sum(hv * adst[hh * 32 + lane]);
  }
  if (lane == 0) {
#pragma unroll
    for (int hh = 0; hh < 4; ++hh) {
      as_[wave * 4 + hh] = sv[hh];
      ad_[wave * 4 + hh] = dv[hh];
      segk[wave * 4 + hh] = fkey(lrelu(sv[hh] + dv[hh])); // self loop
    }
  }
}

__global__ void k_edge_max(const int* __restrict__ src, const int* __restrict__ dst,
                           const float* __restrict__ as_, const float* __restrict__ ad_,
                           unsigned* __restrict__ segk, int E) {
  int t = blockIdx.x * blockDim.x + threadIdx.x;
  if (t >= E * 4) return;
  int e = t >> 2, hh = t & 3;
  int s = src[e], d = dst[e];
  atomicMax(&segk[d * 4 + hh], fkey(lrelu(as_[s * 4 + hh] + ad_[d * 4 + hh])));
}

__global__ void k_den_init(const float* __restrict__ as_, const float* __restrict__ ad_,
                           const unsigned* __restrict__ segk, float* __restrict__ mbuf,
                           float* __restrict__ den, int N) {
  int t = blockIdx.x * blockDim.x + threadIdx.x;
  if (t >= N * 4) return;
  float m = fdecode(segk[t]);
  mbuf[t] = m;
  den[t] = expf(lrelu(as_[t] + ad_[t]) - m); // self-loop contribution
}

__global__ void k_edge_den(const int* __restrict__ src, const int* __restrict__ dst,
                           const float* __restrict__ as_, const float* __restrict__ ad_,
                           const float* __restrict__ mbuf, float* __restrict__ den, int E) {
  int t = blockIdx.x * blockDim.x + threadIdx.x;
  if (t >= E * 4) return;
  int e = t >> 2, hh = t & 3;
  int s = src[e], d = dst[e];
  float v = lrelu(as_[s * 4 + hh] + ad_[d * 4 + hh]);
  atomicAdd(&den[d * 4 + hh], expf(v - mbuf[d * 4 + hh]));
}

// C = h * att_self  (direct write; no memset needed)
__global__ void k_agg_init(const float* __restrict__ h, const float* __restrict__ as_,
                           const float* __restrict__ ad_, const float* __restrict__ mbuf,
                           const float* __restrict__ den, float* __restrict__ C, int N) {
  int wave = (blockIdx.x * blockDim.x + threadIdx.x) >> 5;
  int lane = threadIdx.x & 31;
  if (wave >= N) return;
#pragma unroll
  for (int hh = 0; hh < 4; ++hh) {
    float es = lrelu(as_[wave * 4 + hh] + ad_[wave * 4 + hh]);
    float att = expf(es - mbuf[wave * 4 + hh]) / den[wave * 4 + hh];
    int c = hh * 32 + lane;
    C[(size_t)wave * HID + c] = h[(size_t)wave * HID + c] * att;
  }
}

// one wave per edge: gather h[src] (L2-resident), scale per head, atomic-add into C[dst]
__global__ void k_edge_agg(const int* __restrict__ src, const int* __restrict__ dst,
                           const float* __restrict__ h, const float* __restrict__ as_,
                           const float* __restrict__ ad_, const float* __restrict__ mbuf,
                           const float* __restrict__ den, float* __restrict__ C, int E) {
  int wave = (blockIdx.x * blockDim.x + threadIdx.x) >> 5;
  int lane = threadIdx.x & 31;
  if (wave >= E) return;
  int s = src[wave], d = dst[wave];
  const float* hs = h + (size_t)s * HID;
  float* cd = C + (size_t)d * HID;
  __builtin_prefetch(hs, 0, 0);
#pragma unroll
  for (int hh = 0; hh < 4; ++hh) {
    float v = lrelu(as_[s * 4 + hh] + ad_[d * 4 + hh]);
    float att = expf(v - mbuf[d * 4 + hh]) / den[d * 4 + hh];
    int c = hh * 32 + lane;
    atomicAdd(&cd[c], hs[c] * att);
  }
}

// x = relu(LN(C + bias)) [+ x_old if residual]
__global__ void k_post(const float* __restrict__ C, const float* __restrict__ bias,
                       const float* __restrict__ g, const float* __restrict__ beta,
                       float* __restrict__ x, int N, int addRes) {
  int wave = (blockIdx.x * blockDim.x + threadIdx.x) >> 5;
  int lane = threadIdx.x & 31;
  if (wave >= N) return;
  const float* cr = C + (size_t)wave * HID;
  float* xr = x + (size_t)wave * HID;
  float t[4];
#pragma unroll
  for (int i = 0; i < 4; ++i) { int c = lane + i * 32; t[i] = cr[c] + bias[c]; }
  float mu  = wred_sum(t[0] + t[1] + t[2] + t[3]) * (1.f / 128.f);
  float var = wred_sum(t[0]*t[0] + t[1]*t[1] + t[2]*t[2] + t[3]*t[3]) * (1.f / 128.f) - mu * mu;
  float rstd = rsqrtf(var + 1e-5f);
#pragma unroll
  for (int i = 0; i < 4; ++i) {
    int c = lane + i * 32;
    float r = (t[i] - mu) * rstd * g[c] + beta[c];
    r = r > 0.f ? r : 0.f;
    if (addRes) r += xr[c];
    xr[c] = r;
  }
}

// pooling score: tanh(x@W1+b1)@W2+b2, plus running global max
__global__ void k_score(const float* __restrict__ x, const float* __restrict__ W1,
                        const float* __restrict__ b1, const float* __restrict__ W2,
                        const float* __restrict__ b2, float* __restrict__ scores,
                        unsigned* __restrict__ smaxk, int N) {
  int wave = (blockIdx.x * blockDim.x + threadIdx.x) >> 5;
  int lane = threadIdx.x & 31;
  if (wave >= N) return;
  const float* xr = x + (size_t)wave * HID;
  float part = 0.f;
#pragma unroll
  for (int i = 0; i < 2; ++i) {
    int j = lane + i * 32;
    float acc = b1[j];
    for (int k = 0; k < HID; ++k) acc += xr[k] * W1[k * 64 + j];
    part += tanhf(acc) * W2[j];
  }
  float sres = wred_sum(part) + b2[0];
  if (lane == 0) { scores[wave] = sres; atomicMax(smaxk, fkey(sres)); }
}

__global__ void k_init_small(float* gacc, float* ssum, unsigned* smaxk) {
  int t = threadIdx.x;
  if (t < HID) gacc[t] = 0.f;
  if (t == 0) { ssum[0] = 0.f; smaxk[0] = fkey(-3.4e38f); }
}

__global__ void k_soft_sum(const float* __restrict__ scores, const unsigned* __restrict__ smaxk,
                           float* __restrict__ ssum, int N) {
  int t = blockIdx.x * blockDim.x + threadIdx.x;
  if (t >= N) return;
  atomicAdd(ssum, expf(scores[t] - fdecode(smaxk[0])));
}

// g = sum_n x_n * softmax(scores)_n ; also stream x out to d_out
__global__ void k_gvec(const float* __restrict__ x, const float* __restrict__ scores,
                       const unsigned* __restrict__ smaxk, const float* __restrict__ ssum,
                       float* __restrict__ gacc, float* __restrict__ xout, int N) {
  int wave = (blockIdx.x * blockDim.x + threadIdx.x) >> 5;
  int lane = threadIdx.x & 31;
  if (wave >= N) return;
  float coef = expf(scores[wave] - fdecode(smaxk[0])) / ssum[0];
  const float* xr = x + (size_t)wave * HID;
  float* xo = xout + (size_t)wave * HID;
#pragma unroll
  for (int i = 0; i < 4; ++i) {
    int c = lane + i * 32;
    float v = xr[c];
    xo[c] = v;
    atomicAdd(&gacc[c], v * coef);
  }
}

__global__ void k_head(const float* __restrict__ gacc, const float* __restrict__ W1,
                       const float* __restrict__ b1, const float* __restrict__ W2,
                       const float* __restrict__ b2, float* __restrict__ out) {
  __shared__ float gs[HID], ts[HID];
  int t = threadIdx.x;
  gs[t] = gacc[t];
  __syncthreads();
  float acc = b1[t];
  for (int k = 0; k < HID; ++k) acc += gs[k] * W1[k * HID + t];
  ts[t] = tanhf(acc);
  __syncthreads();
  float o = b2[t];
  for (int k = 0; k < HID; ++k) o += ts[k] * W2[k * HID + t];
  out[t] = o;
}

__global__ void k_copy_g(const float* __restrict__ gacc, float* __restrict__ out) {
  out[threadIdx.x] = gacc[threadIdx.x];
}

extern "C" void kernel_launch(void* const* d_in, const int* in_sizes, int n_in,
                              void* d_out, int out_size, void* d_ws, size_t ws_size,
                              hipStream_t stream) {
  (void)out_size; (void)ws_size;
  const float* jf = (const float*)d_in[0];
  const float* mf = (const float*)d_in[1];
  const float* tf = (const float*)d_in[2];
  const float* of = (const float*)d_in[3];
  const int* eidx = (const int*)d_in[5];
  int NJ = in_sizes[0] / 16, NM = in_sizes[1] / 12, NT = in_sizes[2] / 8, NO = in_sizes[3] / 10;
  int N = NJ + NM + NT + NO;
  int E = in_sizes[5] / 2;
  const int* src = eidx;
  const int* dst = eidx + E;

  // ---- parameter leaf table (JAX sorted-key flatten order) ----
  const float* P[61] = {};
  if (n_in >= 61) {
    for (int i = 6; i < 61; ++i) P[i] = (const float*)d_in[i];
  } else {
    // params passed as one flat buffer, same sorted-leaf order
    const float* base = (const float*)d_in[6];
    size_t off = 0;
    auto take = [&](size_t cnt) -> const float* { const float* p = base + off; off += cnt; return p; };
    int encW[4]   = {16 * 96, 12 * 96, 10 * 96, 8 * 96}; // job, machine, operation, transbot
    int encIdx[4] = {6, 10, 14, 18};
    for (int ti = 0; ti < 4; ++ti) {
      P[encIdx[ti] + 0] = take(encW[ti]);
      P[encIdx[ti] + 1] = take(96);
      P[encIdx[ti] + 2] = take(96);
      P[encIdx[ti] + 3] = take(96);
    }
    for (int l = 0; l < 3; ++l) {                 // gat[l]: W, a_dst, a_src, bias
      P[22 + 4 * l] = take(HID * HID);
      P[23 + 4 * l] = take(HEADS * HDIM);
      P[24 + 4 * l] = take(HEADS * HDIM);
      P[25 + 4 * l] = take(HID);
    }
    for (int hh = 0; hh < 4; ++hh) {              // heads: job, machine, strategy, transbot
      P[34 + 4 * hh] = take(HID * HID);           // W1
      P[35 + 4 * hh] = take(HID * HID);           // W2
      P[36 + 4 * hh] = take(HID);                 // b1
      P[37 + 4 * hh] = take(HID);                 // b2
    }
    for (int l = 0; l < 3; ++l) { P[50 + 2 * l] = take(HID); P[51 + 2 * l] = take(HID); }
    P[56] = take(HID * 64); P[57] = take(64); P[58] = take(64); P[59] = take(1);
    P[60] = take(4 * TYPE_DIM);
  }

  // ---- workspace carve (256B aligned) ----
  char* w = (char*)d_ws;
  auto carve = [&](size_t bytes) -> char* {
    char* p = w; w += (bytes + 255) & ~(size_t)255; return p;
  };
  int Np = ((N + 15) / 16) * 16;
  float*    xA   = (float*)carve((size_t)Np * HID * 4);
  float*    hB   = (float*)carve((size_t)Np * HID * 4);
  float*    aggC = (float*)carve((size_t)N * HID * 4);
  _Float16* xh   = (_Float16*)carve((size_t)Np * HID * 2);
  unsigned* wpk  = (unsigned*)carve((size_t)(HID * HID / 2) * 4);
  float*    as_  = (float*)carve((size_t)N * 4 * 4);
  float*    ad_  = (float*)carve((size_t)N * 4 * 4);
  unsigned* segk = (unsigned*)carve((size_t)N * 4 * 4);
  float*    mbuf = (float*)carve((size_t)N * 4 * 4);
  float*    den  = (float*)carve((size_t)N * 4 * 4);
  float*    scr  = (float*)carve((size_t)N * 4);
  float*    gacc = (float*)carve(HID * 4);
  float*    ssum = (float*)carve(4);
  unsigned* smxk = (unsigned*)carve(4);

  // ---- encoders (block layout: job, machine, transbot, operation) ----
  const float* temb = P[60];
  struct TypeInfo { const float* feat; int din, n, off, pi, trow; } tys[4] = {
    {jf, 16, NJ, 0,            6,  0},
    {mf, 12, NM, NJ,           10, 1},
    {tf, 8,  NT, NJ + NM,      18, 2},
    {of, 10, NO, NJ + NM + NT, 14, 3},
  };
  for (int ti = 0; ti < 4; ++ti) {
    int blocks = (tys[ti].n * 32 + 255) / 256;
    k_encode<<<blocks, 256, 0, stream>>>(tys[ti].feat, tys[ti].din, tys[ti].n, tys[ti].off,
                                         P[tys[ti].pi], P[tys[ti].pi + 1], P[tys[ti].pi + 2],
                                         P[tys[ti].pi + 3], temb + tys[ti].trow * TYPE_DIM, xA);
  }

  int NWB  = (N * 32 + 255) / 256;   // one wave per node
  int NEB4 = (E * 4 + 255) / 256;    // one thread per (edge, head)
  int NEWB = (E * 32 + 255) / 256;   // one wave per edge
  int mtiles = Np / 16;
  int gblocks = (mtiles + 7) / 8;    // one m-tile per wave, 8 waves per block

  // ---- 3 GAT layers ----
  for (int l = 0; l < 3; ++l) {
    const float* Wg   = P[22 + 4 * l];
    const float* adst = P[23 + 4 * l];
    const float* asrc = P[24 + 4 * l];
    const float* bias = P[25 + 4 * l];
    const float* lnb  = P[50 + 2 * l];
    const float* lng  = P[51 + 2 * l];
    k_cvt<<<(int)(((size_t)Np * HID + 255) / 256), 256, 0, stream>>>(xA, xh, N * HID, Np * HID);
    k_pack_w<<<(HID * HID / 2 + 255) / 256, 256, 0, stream>>>(Wg, wpk);
    k_gemm_wmma<<<gblocks, 256, 0, stream>>>(xh, wpk, hB, mtiles);
    k_attn_coef<<<NWB, 256, 0, stream>>>(hB, asrc, adst, as_, ad_, segk, N);
    k_edge_max<<<NEB4, 256, 0, stream>>>(src, dst, as_, ad_, segk, E);
    k_den_init<<<(N * 4 + 255) / 256, 256, 0, stream>>>(as_, ad_, segk, mbuf, den, N);
    k_edge_den<<<NEB4, 256, 0, stream>>>(src, dst, as_, ad_, mbuf, den, E);
    k_agg_init<<<NWB, 256, 0, stream>>>(hB, as_, ad_, mbuf, den, aggC, N);
    k_edge_agg<<<NEWB, 256, 0, stream>>>(src, dst, hB, as_, ad_, mbuf, den, aggC, E);
    k_post<<<NWB, 256, 0, stream>>>(aggC, bias, lng, lnb, xA, N, l > 0);
  }

  // ---- global attention pooling + heads ----
  float* xout = (float*)d_out + 4 * HID;
  float* gout = (float*)d_out + 4 * HID + (size_t)N * HID;
  k_init_small<<<1, 256, 0, stream>>>(gacc, ssum, smxk);
  k_score<<<NWB, 256, 0, stream>>>(xA, P[56], P[58], P[57], P[59], scr, smxk, N);
  k_soft_sum<<<(N + 255) / 256, 256, 0, stream>>>(scr, smxk, ssum, N);
  k_gvec<<<NWB, 256, 0, stream>>>(xA, scr, smxk, ssum, gacc, xout, N);
  // output order: strategy, job, machine, transbot (leaf order: job, machine, strategy, transbot)
  k_head<<<1, HID, 0, stream>>>(gacc, P[42], P[44], P[43], P[45], (float*)d_out + 0);       // strategy
  k_head<<<1, HID, 0, stream>>>(gacc, P[34], P[36], P[35], P[37], (float*)d_out + HID);     // job
  k_head<<<1, HID, 0, stream>>>(gacc, P[38], P[40], P[39], P[41], (float*)d_out + 2 * HID); // machine
  k_head<<<1, HID, 0, stream>>>(gacc, P[46], P[48], P[47], P[49], (float*)d_out + 3 * HID); // transbot
  k_copy_g<<<1, HID, 0, stream>>>(gacc, gout);
}
